// batch_time_InfoNCE_loss_28767690949113
// MI455X (gfx1250) — matrix operand order repaired and checked by reading, not compiled, not run on hardware
//
#include <hip/hip_runtime.h>
#include <hip/hip_bf16.h>

#define BB   128   // batch
#define TT   2048  // time
#define FCC  512   // context features
#define FXX  512   // encoding features
#define NTAU 4
#define TK   128   // K-chunk staged in LDS

typedef __attribute__((ext_vector_type(16))) _Float16 v16h;
typedef __attribute__((ext_vector_type(8)))  float    v8f;
typedef __attribute__((ext_vector_type(4)))  _Float16 v4h;
typedef __attribute__((ext_vector_type(4)))  int      v4i;

// ---------------- CDNA5 async global<->LDS helpers (ASYNCcnt-tracked) ------------------------
#if defined(__AMDGCN__) && __has_builtin(__builtin_amdgcn_global_load_async_to_lds_b128)
#define ASYNC_LDS 1
typedef __attribute__((address_space(1))) v4i g_v4i;   // global int4
typedef __attribute__((address_space(3))) v4i l_v4i;   // LDS int4
#else
#define ASYNC_LDS 0
#endif

__device__ __forceinline__ void async_ld16(_Float16* ldst, const _Float16* gsrc) {
#if ASYNC_LDS
  __builtin_amdgcn_global_load_async_to_lds_b128((g_v4i*)gsrc, (l_v4i*)ldst, 0, 0);
#else
  *(uint4*)ldst = *(const uint4*)gsrc;
#endif
}

__device__ __forceinline__ void async_st16(_Float16* gdst, const _Float16* lsrc) {
#if ASYNC_LDS && __has_builtin(__builtin_amdgcn_global_store_async_from_lds_b128)
  __builtin_amdgcn_global_store_async_from_lds_b128((g_v4i*)gdst, (l_v4i*)lsrc, 0, 0);
#else
  *(uint4*)gdst = *(const uint4*)lsrc;
#endif
}

__device__ __forceinline__ void async_wait0() {
#if ASYNC_LDS
#if __has_builtin(__builtin_amdgcn_s_wait_asynccnt)
  __builtin_amdgcn_s_wait_asynccnt(0);
#else
  asm volatile("s_wait_asynccnt 0" ::: "memory");
#endif
#endif
}

// ---------------- WMMA fragment gathers (16-bit layouts per CDNA5 ISA) -------------
// A 16x32 f16: lane M = lane&15; VGPR v holds K pair {2(v&3)+(v>=4?16:0)} + 8*(lane>>4)
__device__ __forceinline__ v16h frag_a_lds(const _Float16* sm, int row0, int k0) {
  const int lane = threadIdx.x & 31;
  const int l = lane & 15, h = lane >> 4;
  const _Float16* base = sm + (size_t)(row0 + l) * TK + k0 + 8 * h;
  v16h a;
#pragma unroll
  for (int v = 0; v < 8; ++v) {
    const int kp = ((v & 3) * 2) + ((v >> 2) * 16);
    a[2 * v]     = base[kp];
    a[2 * v + 1] = base[kp + 1];
  }
  return a;
}
// B 32x16 f16: lane N = lane&15; VGPR v holds K pair {2(v&3)+(v>=4?16:0)} + 8*(lane>>4)
__device__ __forceinline__ v16h frag_b_lds(const _Float16* sm, int col0, int k0) {
  const int lane = threadIdx.x & 31;
  const int l = lane & 15, h = lane >> 4;
  const _Float16* base = sm + (size_t)(col0 + l) * TK + k0 + 8 * h;
  v16h b;
#pragma unroll
  for (int v = 0; v < 8; ++v) {
    b[2 * v]     = base[2 * v + (v >> 2) * 8];
    b[2 * v + 1] = base[2 * v + 1 + (v >> 2) * 8];
  }
  return b;
}

// Stage a 128x128 fp32 tile from global -> f16 LDS tile (row-major, stride TK)
__device__ __forceinline__ void stage_f32_tile(_Float16* dst, const float* src_base,
                                               size_t row_stride_f) {
  const int tid = threadIdx.x;
#pragma unroll
  for (int it = 0; it < 16; ++it) {
    const int idx = tid + it * 256;   // 0..4095, 32 float4 per row
    const int row = idx >> 5;
    const int c4  = idx & 31;
    const float4 v = *(const float4*)(src_base + (size_t)row * row_stride_f + c4 * 4);
    v4h hv = {(_Float16)v.x, (_Float16)v.y, (_Float16)v.z, (_Float16)v.w};
    *(v4h*)(dst + row * TK + c4 * 4) = hv;
  }
}

// 128x128 += A(128xTK) * B(128xTK)^T ; wave owns rows [wid*16, wid*16+16).
// Preload all 8 B fragments per K-step so the 8 independent WMMAs issue back-to-back.
__device__ __forceinline__ void mma_tiles(const _Float16* sA, const _Float16* sB,
                                          v8f* acc, int wid) {
#pragma unroll
  for (int kk = 0; kk < TK; kk += 32) {
    const v16h a = frag_a_lds(sA, wid * 16, kk);
    v16h b[8];
#pragma unroll
    for (int nb = 0; nb < 8; ++nb) b[nb] = frag_b_lds(sB, nb * 16, kk);
#pragma unroll
    for (int nb = 0; nb < 8; ++nb)
      acc[nb] = __builtin_amdgcn_wmma_f32_16x16x32_f16(
          false, a, false, b[nb], (short)0, acc[nb], false, false);
  }
}

// ---------------- Kernel 1: projection  Xt[k,tc,j,:] = X[j, t+tau, :] @ W[k]^T + b[k] --------
__global__ __launch_bounds__(256) void k_proj(const float* __restrict__ X,
                                              const float* __restrict__ W,
                                              const float* __restrict__ bias,
                                              _Float16* __restrict__ xt,
                                              int t0, int tchunk) {
  const int gb = blockIdx.x;        // output col block (4 x 128)
  const int tc = blockIdx.y;        // time within chunk
  const int k  = blockIdx.z;        // tau index
  const int tau = 1 << k;
  const int tg = t0 + tc;
  if (tg >= TT - tau) return;
  const int xtime = tg + tau;
  const int g0 = gb * 128;

  __shared__ __align__(16) _Float16 sX[BB * TK];
  __shared__ __align__(16) _Float16 sW[BB * TK];

  const int tid  = threadIdx.x;
  const int wid  = tid >> 5;
  const int lane = tid & 31;
  const int l = lane & 15, h = lane >> 4;

  v8f acc[8] = {};
  for (int f0 = 0; f0 < FXX; f0 += TK) {
    if (f0 + TK < FXX)   // hint next K-chunk toward the caches
      __builtin_prefetch(X + (size_t)xtime * FXX + f0 + TK, 0, 1);
    stage_f32_tile(sX, X + (size_t)xtime * FXX + f0, (size_t)TT * FXX);
    stage_f32_tile(sW, W + ((size_t)k * FCC + g0) * FXX + f0, (size_t)FXX);
    __syncthreads();
    mma_tiles(sX, sW, acc, wid);
    __syncthreads();
  }

  // Repack output tile through LDS so global stores are coalesced 128B async lines.
  _Float16* so = sX;   // safe: trailing __syncthreads() above
#pragma unroll
  for (int nb = 0; nb < 8; ++nb) {
    const int gc = nb * 16 + l;
    const float bv = bias[k * FCC + g0 + gc];
#pragma unroll
    for (int r = 0; r < 8; ++r) {
      const int j = wid * 16 + 8 * h + r;   // C/D layout: VGPR r holds row r + 8h
      so[j * TK + gc] = (_Float16)(acc[nb][r] + bv);
    }
  }
  __syncthreads();
  const size_t obase = (((size_t)k * tchunk + tc) * BB) * FCC + g0;
#pragma unroll
  for (int it = 0; it < 8; ++it) {
    const int idx = tid + it * 256;        // 0..2047, 16 x 16B per row
    const int row = idx >> 4;
    const int q   = idx & 15;
    async_st16(xt + obase + (size_t)row * FCC + q * 8, so + row * TK + q * 8);
  }
  async_wait0();
}

// ---------------- Kernel 2: scores Z = C_t @ Xt_t^T, fused logsumexp + diag ------------------
__global__ __launch_bounds__(256) void k_score(const float* __restrict__ C,
                                               const _Float16* __restrict__ xt,
                                               double* __restrict__ accd,
                                               int t0, int tchunk) {
  const int tc = blockIdx.x;
  const int k  = blockIdx.y;
  const int tau = 1 << k;
  const int tg = t0 + tc;
  if (tg >= TT - tau) return;

  __shared__ __align__(16) _Float16 sC[BB * TK];
  __shared__ __align__(16) _Float16 sXt[BB * TK];

  const int tid  = threadIdx.x;
  const int wid  = tid >> 5;
  const int lane = tid & 31;
  const int l = lane & 15, h = lane >> 4;

  v8f acc[8] = {};
  const _Float16* xbase = xt + (((size_t)k * tchunk + tc) * BB) * FCC;
  for (int f0 = 0; f0 < FCC; f0 += TK) {
    // Xt tile (already f16, L2-resident): async global->LDS, overlaps with C convert below.
#pragma unroll
    for (int it = 0; it < 8; ++it) {
      const int idx = tid + it * 256;       // 0..2047, 16 uint4 per row
      const int row = idx >> 4;
      const int q   = idx & 15;
      async_ld16(sXt + row * TK + q * 8, xbase + (size_t)row * FCC + f0 + q * 8);
    }
    stage_f32_tile(sC, C + (size_t)tg * FCC + f0, (size_t)TT * FCC);
    async_wait0();
    __syncthreads();
    mma_tiles(sC, sXt, acc, wid);
    __syncthreads();
  }

  // Each wave owns rows i = wid*16 + 8h + r over ALL 128 cols (full lse per row in-wave).
  float wave_part = 0.f;
#pragma unroll
  for (int r = 0; r < 8; ++r) {
    float m = -3.0e38f;
#pragma unroll
    for (int nb = 0; nb < 8; ++nb) m = fmaxf(m, acc[nb][r]);
#pragma unroll
    for (int s = 1; s <= 8; s <<= 1) m = fmaxf(m, __shfl_xor(m, s, 32));
    float e = 0.f;
#pragma unroll
    for (int nb = 0; nb < 8; ++nb) e += __expf(acc[nb][r] - m);
#pragma unroll
    for (int s = 1; s <= 8; s <<= 1) e += __shfl_xor(e, s, 32);
    // diagonal Z[i,i]: nb==wid, lane l == (i & 15) == 8h + r within same half
    float dv = (l == 8 * h + r) ? acc[wid][r] : 0.f;
#pragma unroll
    for (int s = 1; s <= 8; s <<= 1) dv += __shfl_xor(dv, s, 32);
    if (l == 0) wave_part += dv - (__logf(e) + m);
  }

  __syncthreads();
  float* red = (float*)sC;                  // reuse LDS after last barrier
  if (l == 0) red[wid * 2 + h] = wave_part;
  __syncthreads();
  if (tid == 0) {
    float tot = 0.f;
#pragma unroll
    for (int q = 0; q < 16; ++q) tot += red[q];
    atomicAdd(&accd[k], (double)tot);
  }
}

__global__ void k_zero(double* accd) {
  if (threadIdx.x < NTAU) accd[threadIdx.x] = 0.0;
}

__global__ void k_final(const double* __restrict__ accd, float* __restrict__ out) {
  const int k = threadIdx.x;
  if (k < NTAU) {
    const int tau = 1 << k;
    out[k] = (float)(accd[k] / ((double)BB * (double)(TT - tau)));
  }
}

extern "C" void kernel_launch(void* const* d_in, const int* in_sizes, int n_in,
                              void* d_out, int out_size, void* d_ws, size_t ws_size,
                              hipStream_t stream) {
  const float* C    = (const float*)d_in[0];   // [B, T, FC]
  const float* X    = (const float*)d_in[1];   // [B, T, FX]
  const float* W    = (const float*)d_in[2];   // [4, FC, FX]
  const float* bias = (const float*)d_in[3];   // [4, FC]

  double*   accd = (double*)d_ws;
  _Float16* xt   = (_Float16*)((char*)d_ws + 256);

  // Pick the largest time-chunk whose Xt staging fits workspace (target: stay L2-resident).
  const size_t per_t = (size_t)NTAU * BB * FCC * sizeof(_Float16);  // 512 KB per time step
  int tchunk = 128;
  while (tchunk > 8 && 256 + (size_t)tchunk * per_t > ws_size) tchunk >>= 1;

  k_zero<<<1, 32, 0, stream>>>(accd);
  for (int t0 = 0; t0 < TT; t0 += tchunk) {
    const int tcnt = (TT - t0 < tchunk) ? (TT - t0) : tchunk;
    k_proj <<<dim3(4, tcnt, NTAU), 256, 0, stream>>>(X, W, bias, xt, t0, tchunk);
    k_score<<<dim3(tcnt, NTAU),    256, 0, stream>>>(C, xt, accd, t0, tchunk);
  }
  k_final<<<1, 32, 0, stream>>>(accd, (float*)d_out);
}
